// Attention_3435973837469
// MI455X (gfx1250) — compile-verified, matrix operand order checked by reading, not run on hardware
//
#include <hip/hip_runtime.h>

// ---------------------------------------------------------------------------
// MI455X (gfx1250) attention with pair biases.
// Roofline: bias1+bias2 = 0.54 GB streamed once @ 23.3 TB/s ~= 23 us floor;
// ~20 GF of math << bf16 WMMA capability => memory bound. Design:
//  - flash streaming over K in 128-wide chunks; biases read once, NT hints,
//    and fed to the score WMMA as the C operand (D = QK^T + bias, fused).
//  - v_wmma_f32_16x16x32_bf16 (WMMA K = 32 = head dim).
//  - head-major K / transposed V from projection epilogues so attention tile
//    staging is contiguous b128 blocks -> global_load_async_to_lds_b128,
//    double-buffered on ASYNCcnt to overlap with compute.
//  - softmax row-sums via WMMA against an all-ones fragment (no cross-lane
//    reduction); row-max via pipelined independent ds_bpermutes.
// ---------------------------------------------------------------------------

typedef __attribute__((ext_vector_type(16))) __bf16 v16bf;
typedef __attribute__((ext_vector_type(8)))  __bf16 v8bf;
typedef __attribute__((ext_vector_type(8)))  float  v8f;

#define B_  2
#define QL  2048
#define KL  2048
#define NH  8
#define HD  32     // head dim C (== WMMA K for bf16 x32)
#define HC  256    // H*C == CQ == CK == CV == CO
#define KC  128    // k positions streamed per chunk

// Build a 16x32-bf16 A/B fragment (8 VGPRs x 2 bf16) from two contiguous
// 16-byte chunks, per the CDNA5 ISA 16-bit matrix layout:
//   lane<16 : K = {0..7, 16..23};  lane>=16 : K = {8..15, 24..31}
static __device__ inline v16bf make_frag(const __bf16* p0, const __bf16* p1) {
  v8bf lo = *(const v8bf*)p0;
  v8bf hi = *(const v8bf*)p1;
  v16bf f;
#pragma unroll
  for (int e = 0; e < 8; ++e) { f[e] = lo[e]; f[e + 8] = hi[e]; }
  return f;
}

// 32-bit LDS offset of a shared-memory pointer (for async-to-LDS VDST).
typedef __attribute__((address_space(3))) const void* lds_cptr_t;
static __device__ inline unsigned lds_addr(const void* p) {
  return (unsigned)(unsigned long long)(lds_cptr_t)p;
}

// Async 16-byte global -> LDS copy (gfx1250 async-to-LDS path, ASYNCcnt).
static __device__ inline void async_copy16(unsigned lds_off, const void* g) {
  asm volatile("global_load_async_to_lds_b128 %0, %1, off"
               :: "v"(lds_off), "v"((unsigned long long)(size_t)g)
               : "memory");
}
static __device__ inline void wait_asynccnt0() {
  asm volatile("s_wait_asynccnt 0x0" ::: "memory");
}

// ---------------------------------------------------------------------------
// GEMM out = A[M,256] * W[256,256] with fused, compile-time epilogue.
// Block: 256 threads (8 waves); block tile 16(M) x 128(N); wave tile 16x16.
// MODE 0: q  -> head-major bf16, * C^-0.5        [b,h,q,c]
// MODE 1: k  -> head-major bf16                  [b,h,k,c]
// MODE 2: v  -> transposed bf16                  [b,h,c,k]
// MODE 3: g  -> f32 sigmoid(x + bias)            [b,q,h*c]
// MODE 4: out-> f32 x + bias                     [b,q,h*c]
// ---------------------------------------------------------------------------
template <int MODE, bool ABF16>
__global__ __launch_bounds__(256)
void gemm16(const void* __restrict__ Ain, const float* __restrict__ W,
            const float* __restrict__ bias, void* __restrict__ out,
            float scale)
{
  constexpr int N = 256, Kd = 256;
  __shared__ __bf16 alds[16 * 32];
  __shared__ __bf16 blds[128 * 32];   // staged transposed: [n][k]
  const int t = threadIdx.x;
  const int wave = t >> 5, lane = t & 31;
  const int hi = (lane >> 4) & 1, ln = lane & 15;
  const int m0 = blockIdx.x * 16;
  const int n0 = blockIdx.y * 128;

  v8f acc = {};
  for (int k0 = 0; k0 < Kd; k0 += 32) {
    __syncthreads();
#pragma unroll
    for (int e = 0; e < 2; ++e) {
      int idx = t * 2 + e;
      int r = idx >> 5, kk = idx & 31;
      size_t src = (size_t)(m0 + r) * Kd + (k0 + kk);
      float v = ABF16 ? (float)((const __bf16*)Ain)[src]
                      : ((const float*)Ain)[src];
      alds[r * 32 + kk] = (__bf16)v;
    }
    {
      int base = t * 16;
      int kk = base >> 7;
      int nn = base & 127;
#pragma unroll
      for (int e = 0; e < 16; ++e) {
        float v = W[(size_t)(k0 + kk) * N + (n0 + nn + e)];
        blds[(nn + e) * 32 + kk] = (__bf16)v;
      }
    }
    __syncthreads();
    v16bf af = make_frag(&alds[ln * 32 + 8 * hi], &alds[ln * 32 + 16 + 8 * hi]);
    const int nb = wave * 16 + ln;
    v16bf bfr = make_frag(&blds[nb * 32 + 8 * hi], &blds[nb * 32 + 16 + 8 * hi]);
    acc = __builtin_amdgcn_wmma_f32_16x16x32_bf16(false, af, false, bfr,
                                                  (short)0, acc, false, false);
  }

  const int ncol = n0 + wave * 16 + ln;
  const int h = ncol >> 5, c = ncol & 31;
#pragma unroll
  for (int j = 0; j < 8; ++j) {
    int row = m0 + hi * 8 + j;          // global M index = b*L + pos
    int b = row >> 11, pos = row & (QL - 1);
    float v = acc[j];
    if (MODE == 0) {
      size_t o = (((size_t)(b * NH + h) * QL + pos) << 5) + c;
      ((__bf16*)out)[o] = (__bf16)(v * scale);
    } else if (MODE == 1) {
      size_t o = (((size_t)(b * NH + h) * KL + pos) << 5) + c;
      ((__bf16*)out)[o] = (__bf16)v;
    } else if (MODE == 2) {
      size_t o = ((size_t)(b * NH + h) * HD + c) * KL + pos;
      ((__bf16*)out)[o] = (__bf16)v;
    } else if (MODE == 3) {
      float x = v + bias[ncol];
      ((float*)out)[(size_t)row * N + ncol] = 1.0f / (1.0f + __expf(-x));
    } else {
      ((float*)out)[(size_t)row * N + ncol] = v + bias[ncol];
    }
  }
}

// ---------------------------------------------------------------------------
// Flash attention. Block = 256 threads (8 waves) handles one (b,h) and 128 q
// rows (16/wave). K streamed in 128-wide chunks, double-buffered async-to-LDS
// staging; bias1/bias2 read once (NT) and fused into the score WMMA as C.
// ---------------------------------------------------------------------------
__global__ __launch_bounds__(256)
void attn_kernel(const __bf16* __restrict__ qH, const __bf16* __restrict__ kH,
                 const __bf16* __restrict__ vT, const float* __restrict__ bias1,
                 const float* __restrict__ bias2, const float* __restrict__ gate,
                 __bf16* __restrict__ o16)
{
  __shared__ __bf16 klds[2][KC * 32];       // [buf][kpos][c]  (c contiguous)
  __shared__ __bf16 vlds[2][32 * KC];       // [buf][c][kpos]  (k contiguous)
  __shared__ __bf16 slds[8][16 * KC];       // per-wave P scratch

  const int t = threadIdx.x, wave = t >> 5, lane = t & 31;
  const int hi = (lane >> 4) & 1, ln = lane & 15;
  const int b = blockIdx.z, h = blockIdx.y;
  const int bh = b * NH + h;
  const int q0 = blockIdx.x * 128 + wave * 16;

  // Q fragment (A layout): row = lane&15, K(=C) split per half-wave.
  const __bf16* qrow = qH + (((size_t)bh * QL + q0 + ln) << 5);
  v16bf qf = make_frag(qrow + 8 * hi, qrow + 16 + 8 * hi);

  // Constant all-ones B fragment for row-sum WMMAs.
  v16bf onesb;
#pragma unroll
  for (int e = 0; e < 16; ++e) onesb[e] = (__bf16)1.0f;

  float m[8];
#pragma unroll
  for (int j = 0; j < 8; ++j) m[j] = -1e30f;
  v8f O0 = {}, O1 = {}, Lacc = {};
  __bf16* sw = &slds[wave][0];

  // Staging: each thread does 2 k-copies + 2 v-copies (16B each) per chunk.
  const unsigned kd[2] = { lds_addr(&klds[0][t * 8]), lds_addr(&klds[1][t * 8]) };
  const int vc = t >> 3, vk0 = (t & 7) * 8;
  const unsigned vd[2] = { lds_addr(&vlds[0][vc * KC + vk0]),
                           lds_addr(&vlds[1][vc * KC + vk0]) };
  const __bf16* kbase = kH + (((size_t)bh * KL) << 5) + t * 8;
  const __bf16* vbase = vT + ((size_t)bh * HD + vc) * KL + vk0;

  auto issue_chunk = [&](int kc, int s) {
    async_copy16(kd[s],        kbase + (size_t)kc * HD);
    async_copy16(kd[s] + 4096, kbase + (size_t)kc * HD + 2048);
    async_copy16(vd[s],        vbase + kc);
    async_copy16(vd[s] + 128,  vbase + kc + 64);
  };

  issue_chunk(0, 0);

  constexpr int NCH = KL / KC;   // 16
  for (int i = 0; i < NCH; ++i) {
    const int kc = i * KC;
    wait_asynccnt0();            // this chunk's async tiles landed
    __syncthreads();             // visible to all waves; prev bufs consumed
    if (i + 1 < NCH) issue_chunk(kc + KC, (i + 1) & 1);
    const __bf16* kb = klds[i & 1];
    const __bf16* vb = vlds[i & 1];

    // Scores with fused pair biases: S = Q K^T + (bias1 + bias2).
    // Bias loads are the bandwidth-dominant stream: read exactly once, NT,
    // accumulated in C layout and passed as the WMMA C operand.
    const size_t bb = (size_t)bh * QL * KL;
    v8f S[8];
#pragma unroll
    for (int nc = 0; nc < 8; ++nc) {
      int col = kc + nc * 16 + ln;
      v8f bia;
#pragma unroll
      for (int j = 0; j < 8; ++j) {
        size_t idx = bb + (size_t)(q0 + hi * 8 + j) * KL + col;
        bia[j] = __builtin_nontemporal_load(bias1 + idx)
               + __builtin_nontemporal_load(bias2 + idx);
      }
      int n = nc * 16 + ln;
      v16bf kf = make_frag(&kb[n * 32 + 8 * hi], &kb[n * 32 + 16 + 8 * hi]);
      S[nc] = __builtin_amdgcn_wmma_f32_16x16x32_bf16(false, qf, false, kf,
                                                      (short)0, bia, false, false);
    }

    // Row max: local max3 tree, then pipelined butterflies (8 independent
    // bpermutes per stage instead of 8 serialized chains).
    float rm[8];
#pragma unroll
    for (int j = 0; j < 8; ++j) {
      float a0 = fmaxf(fmaxf(S[0][j], S[1][j]), fmaxf(S[2][j], S[3][j]));
      float a1 = fmaxf(fmaxf(S[4][j], S[5][j]), fmaxf(S[6][j], S[7][j]));
      rm[j] = fmaxf(a0, a1);
    }
#pragma unroll
    for (int msk = 8; msk >= 1; msk >>= 1) {
      float tmp[8];
#pragma unroll
      for (int j = 0; j < 8; ++j) tmp[j] = __shfl_xor(rm[j], msk, 32);
#pragma unroll
      for (int j = 0; j < 8; ++j) rm[j] = fmaxf(rm[j], tmp[j]);
    }

    // Online rescale + exponentials (unnormalized P).
#pragma unroll
    for (int j = 0; j < 8; ++j) {
      float mn = fmaxf(m[j], rm[j]);
      float alpha = __expf(m[j] - mn);
      m[j] = mn;
#pragma unroll
      for (int nc = 0; nc < 8; ++nc) S[nc][j] = __expf(S[nc][j] - mn);
      O0[j] *= alpha; O1[j] *= alpha; Lacc[j] *= alpha;
    }

    // P (C layout) -> bf16 -> wave-private LDS -> A-layout fragments.
#pragma unroll
    for (int nc = 0; nc < 8; ++nc)
#pragma unroll
      for (int j = 0; j < 8; ++j)
        sw[(hi * 8 + j) * KC + nc * 16 + ln] = (__bf16)S[nc][j];

    // PV + row-sum: per 32-k slab: 2 output-column WMMAs + 1 ones-WMMA.
#pragma unroll
    for (int ck = 0; ck < KC; ck += 32) {
      v16bf pf  = make_frag(&sw[ln * KC + ck + 8 * hi],
                            &sw[ln * KC + ck + 16 + 8 * hi]);
      v16bf vf0 = make_frag(&vb[ln * KC + ck + 8 * hi],
                            &vb[ln * KC + ck + 16 + 8 * hi]);
      v16bf vf1 = make_frag(&vb[(16 + ln) * KC + ck + 8 * hi],
                            &vb[(16 + ln) * KC + ck + 16 + 8 * hi]);
      O0 = __builtin_amdgcn_wmma_f32_16x16x32_bf16(false, pf, false, vf0,
                                                   (short)0, O0, false, false);
      O1 = __builtin_amdgcn_wmma_f32_16x16x32_bf16(false, pf, false, vf1,
                                                   (short)0, O1, false, false);
      Lacc = __builtin_amdgcn_wmma_f32_16x16x32_bf16(false, pf, false, onesb,
                                                     (short)0, Lacc, false, false);
    }
  }

  // Epilogue: normalize by WMMA row sums, apply gate, store bf16.
#pragma unroll
  for (int j = 0; j < 8; ++j) {
    int row = q0 + hi * 8 + j;
    float inv = 1.0f / Lacc[j];
    size_t base = (size_t)(b * QL + row) * HC + h * HD;
    float g0 = gate[base + ln];
    float g1 = gate[base + 16 + ln];
    o16[base + ln]      = (__bf16)(O0[j] * inv * g0);
    o16[base + 16 + ln] = (__bf16)(O1[j] * inv * g1);
  }
}

// ---------------------------------------------------------------------------
extern "C" void kernel_launch(void* const* d_in, const int* in_sizes, int n_in,
                              void* d_out, int out_size, void* d_ws, size_t ws_size,
                              hipStream_t stream) {
  (void)in_sizes; (void)n_in; (void)out_size; (void)ws_size;
  const float* q_x   = (const float*)d_in[0];
  const float* kv_x  = (const float*)d_in[1];
  const float* bias1 = (const float*)d_in[2];
  const float* bias2 = (const float*)d_in[3];
  const float* Wq    = (const float*)d_in[4];
  const float* Wk    = (const float*)d_in[5];
  const float* Wv    = (const float*)d_in[6];
  const float* Wg    = (const float*)d_in[7];
  const float* bg    = (const float*)d_in[8];
  const float* Wo    = (const float*)d_in[9];
  const float* bo    = (const float*)d_in[10];
  float* out = (float*)d_out;

  char* ws = (char*)d_ws;
  __bf16* qH  = (__bf16*)(ws);                        // 2 MB  [b,h,q,c]
  __bf16* kH  = (__bf16*)(ws + ((size_t)2 << 20));    // 2 MB  [b,h,k,c]
  __bf16* vT  = (__bf16*)(ws + ((size_t)4 << 20));    // 2 MB  [b,h,c,k]
  float*  gte = (float*) (ws + ((size_t)6 << 20));    // 4 MB  [b,q,h*c]
  __bf16* o16 = (__bf16*)(ws + ((size_t)10 << 20));   // 2 MB  [b,q,h*c]

  const int M = B_ * QL;   // 4096
  dim3 blk(256);
  dim3 gproj(M / 16, HC / 128);   // (256, 2)
  const float qscale = 0.17677669529663687f;  // 32^-0.5

  gemm16<0, false><<<gproj, blk, 0, stream>>>(q_x,  Wq, nullptr, qH,  qscale);
  gemm16<1, false><<<gproj, blk, 0, stream>>>(kv_x, Wk, nullptr, kH,  1.0f);
  gemm16<2, false><<<gproj, blk, 0, stream>>>(kv_x, Wv, nullptr, vT,  1.0f);
  gemm16<3, false><<<gproj, blk, 0, stream>>>(q_x,  Wg, bg,      gte, 1.0f);

  dim3 gattn(QL / 128, NH, B_);   // (16, 8, 2)
  attn_kernel<<<gattn, blk, 0, stream>>>(qH, kH, vT, bias1, bias2, gte, o16);

  gemm16<4, true><<<gproj, blk, 0, stream>>>(o16, Wo, bo, out, 1.0f);
}
